// CausalSelfAttentionWithALiBi_75788992905418
// MI455X (gfx1250) — matrix operand order repaired and checked by reading, not compiled, not run on hardware
//
#include <hip/hip_runtime.h>

// ---------------------------------------------------------------------------
// CausalSelfAttentionWithALiBi on gfx1250 (MI455X), f16 WMMA / f32 accumulate.
// B=2, S=2048, E=1024, H=16, D=64.
// Pipeline: cvt/transpose -> kqv GEMM (WMMA) -> flash attention (WMMA)
//           -> proj GEMM (WMMA).
// Round-3: register double-buffered GEMM mainloop (global loads for tile k+1
// overlap the 8 WMMAs of tile k); attention unchanged from round 2 (compiler
// already software-pipelines it).
// ---------------------------------------------------------------------------

typedef __attribute__((ext_vector_type(16))) _Float16 v16h;
typedef __attribute__((ext_vector_type(8)))  float    v8f;

#define B_   2
#define S_   2048
#define E_   1024
#define H_   16
#define D_   64
#define MR_  (B_ * S_)      // 4096 rows
#define NKQV (3 * E_)       // 3072

union Frag16 { uint4 u4[2]; v16h h; };

__device__ __forceinline__ v8f wmma_f16(v16h a, v16h b, v8f c) {
  return __builtin_amdgcn_wmma_f32_16x16x32_f16(
      /*neg_a=*/false, a, /*neg_b=*/false, b,
      /*c_mod=*/(short)0, c, /*reuse_a=*/false, /*reuse_b=*/false);
}

// DPP16 rotate-right by N within each 16-lane row (VALU, no wait); ror 1,2,4,8
// reduction yields the full 16-lane result in every lane, never crossing the
// 16-lane half boundary.
template <int N>
__device__ __forceinline__ float row_ror(float v) {
  int iv = __float_as_int(v);
  int r = __builtin_amdgcn_update_dpp(iv, iv, 0x120 + N, 0xf, 0xf, true);
  return __int_as_float(r);
}
__device__ __forceinline__ float rowmax16(float v) {
  v = fmaxf(v, row_ror<1>(v));
  v = fmaxf(v, row_ror<2>(v));
  v = fmaxf(v, row_ror<4>(v));
  v = fmaxf(v, row_ror<8>(v));
  return v;
}
__device__ __forceinline__ float rowsum16(float v) {
  v += row_ror<1>(v);
  v += row_ror<2>(v);
  v += row_ror<4>(v);
  v += row_ror<8>(v);
  return v;
}

// A fragment (16x32 f16, row-major source, stride in halves).
__device__ __forceinline__ v16h load_a_frag(const _Float16* base, int stride_h, int lane) {
  Frag16 f;
  const _Float16* r = base + (size_t)(lane & 15) * stride_h + ((lane >> 4) << 3);
  f.u4[0] = *(const uint4*)(r);
  f.u4[1] = *(const uint4*)(r + 16);
  return f.h;
}

// B fragment (32x16 f16) from a K-contiguous "Bt" source.
__device__ __forceinline__ v16h load_b_frag(const _Float16* base, int stride_h, int lane) {
  Frag16 f;
  const _Float16* r = base + (size_t)(lane & 15) * stride_h + ((lane >> 4) << 3);
  f.u4[0] = *(const uint4*)(r);
  f.u4[1] = *(const uint4*)(r + 16);
  return f.h;
}

// ---------------------------------------------------------------------------
__global__ __launch_bounds__(256) void cvt_f32_f16(const float* __restrict__ in,
                                                   _Float16* __restrict__ out, int n) {
  int i = blockIdx.x * 256 + threadIdx.x;
  if (i < n) out[i] = (_Float16)in[i];
}

__global__ __launch_bounds__(256) void transpose_cvt(const float* __restrict__ in,
                                                     _Float16* __restrict__ out,
                                                     int K, int N) {
  __shared__ float tile[32][33];
  int nb = blockIdx.x * 32, kb = blockIdx.y * 32;
  #pragma unroll
  for (int i = threadIdx.y; i < 32; i += 8)
    tile[i][threadIdx.x] = in[(size_t)(kb + i) * N + nb + threadIdx.x];
  __syncthreads();
  #pragma unroll
  for (int i = threadIdx.y; i < 32; i += 8)
    out[(size_t)(nb + i) * K + kb + threadIdx.x] = (_Float16)tile[threadIdx.x][i];
}

// ---------------------------------------------------------------------------
// C[M][N] = A[M][K] * Bt[N][K]^T + bias. BM=128, BN=64, BK=64; 8 waves, 32x32 each.
// Register double-buffered: tile k+1 is loaded into VGPRs while tile k computes.
template <int MODE>
__global__ __launch_bounds__(256) void gemm_f16(
    const _Float16* __restrict__ A, const _Float16* __restrict__ Bt,
    const float* __restrict__ bias, int M, int N, int K,
    float* __restrict__ outf,
    _Float16* __restrict__ Kh, _Float16* __restrict__ Qh, _Float16* __restrict__ Vt) {
  __shared__ _Float16 sA[128 * 72];  // 144B row stride: 16B aligned, conflict-free
  __shared__ _Float16 sB[64 * 72];
  const int tid = threadIdx.x, lane = tid & 31, wave = tid >> 5;
  const int m0 = blockIdx.y * 128, n0 = blockIdx.x * 64;
  const int wm0 = (wave >> 1) * 32, wn0 = (wave & 1) * 32;
  const int arow = tid >> 1, aseg = tid & 1;  // A: 2 threads/row, 32 halves each
  const int brow = tid >> 2, bseg = tid & 3;  // B: 4 threads/row, 16 halves each

  const uint4* asrc = (const uint4*)(A + (size_t)(m0 + arow) * K + aseg * 32);
  const uint4* bsrc = (const uint4*)(Bt + (size_t)(n0 + brow) * K + bseg * 16);
  uint4* adst = (uint4*)(sA + arow * 72 + aseg * 32);
  uint4* bdst = (uint4*)(sB + brow * 72 + bseg * 16);

  uint4 ra[4], rb[2];
  // Preload tile k0 = 0 (global addresses advance by 64 halves = 8 uint4 / tile).
  ra[0] = asrc[0]; ra[1] = asrc[1]; ra[2] = asrc[2]; ra[3] = asrc[3];
  rb[0] = bsrc[0]; rb[1] = bsrc[1];

  v8f acc[2][2] = {};
  for (int k0 = 0; k0 < K; k0 += 64) {
    adst[0] = ra[0]; adst[1] = ra[1]; adst[2] = ra[2]; adst[3] = ra[3];
    bdst[0] = rb[0]; bdst[1] = rb[1];
    __syncthreads();
    // Issue next tile's global loads; latency covered by the 8 WMMAs below.
    const int knext = (k0 + 64 < K) ? (k0 + 64) : k0;  // clamped (redundant on last)
    {
      const uint4* an = asrc + (knext >> 3);  // 64 halves = 8 uint4
      const uint4* bn = bsrc + (knext >> 3);
      ra[0] = an[0]; ra[1] = an[1]; ra[2] = an[2]; ra[3] = an[3];
      rb[0] = bn[0]; rb[1] = bn[1];
    }
    #pragma unroll
    for (int ks = 0; ks < 2; ++ks) {
      v16h a0 = load_a_frag(sA + (wm0 + 0) * 72 + ks * 32, 72, lane);
      v16h a1 = load_a_frag(sA + (wm0 + 16) * 72 + ks * 32, 72, lane);
      v16h b0 = load_b_frag(sB + (wn0 + 0) * 72 + ks * 32, 72, lane);
      v16h b1 = load_b_frag(sB + (wn0 + 16) * 72 + ks * 32, 72, lane);
      acc[0][0] = wmma_f16(a0, b0, acc[0][0]);
      acc[0][1] = wmma_f16(a0, b1, acc[0][1]);
      acc[1][0] = wmma_f16(a1, b0, acc[1][0]);
      acc[1][1] = wmma_f16(a1, b1, acc[1][1]);
    }
    __syncthreads();
  }

  // Epilogue. C layout: m = r + 8*(lane/16), n = lane%16 within each 16x16 frag.
  #pragma unroll
  for (int mi = 0; mi < 2; ++mi)
    #pragma unroll
    for (int ni = 0; ni < 2; ++ni)
      #pragma unroll
      for (int r = 0; r < 8; ++r) {
        int m = m0 + wm0 + mi * 16 + r + ((lane >> 4) << 3);
        int n = n0 + wn0 + ni * 16 + (lane & 15);
        float v = acc[mi][ni][r] + bias[n];
        if (MODE == 0) {
          outf[(size_t)m * N + n] = v;
        } else {
          int which = n >> 10, hd = n & 1023;      // split order: k, q, v
          int h = hd >> 6, d = hd & 63;
          int b = m >> 11, s = m & 2047;
          size_t qk = ((size_t)(b * H_ + h) * S_ + s) * D_ + d;
          if (which == 0)       Kh[qk] = (_Float16)v;
          else if (which == 1)  Qh[qk] = (_Float16)(v * 0.125f);  // fold 1/sqrt(D)
          else                  Vt[((size_t)(b * H_ + h) * D_ + d) * S_ + s] = (_Float16)v;
        }
      }
}

// ---------------------------------------------------------------------------
// Flash attention with ALiBi. One wave per (b, h, 16-row q tile); key blocks of 32.
// K fragments are rotated: block kb+1 is prefetched while block kb computes.
__global__ __launch_bounds__(256) void attn_kernel(
    const _Float16* __restrict__ Qh, const _Float16* __restrict__ Kh,
    const _Float16* __restrict__ Vt, const float* __restrict__ slopes,
    _Float16* __restrict__ attnh) {
  __shared__ _Float16 pshm[8][16 * 40];  // per-wave P scratch (16x32, pad->40)
  const int lane = threadIdx.x & 31;
  const int w = threadIdx.x >> 5;
  const int wave = blockIdx.x * 8 + w;
  const int qt = wave & 127;           // S/16 = 128 tiles
  const int h  = (wave >> 7) & 15;
  const int b  = wave >> 11;
  const int q0 = qt << 4;
  const int jl = lane & 15, half = lane >> 4;

  const _Float16* Qp = Qh + (size_t)(b * H_ + h) * S_ * D_;
  const _Float16* Kp = Kh + (size_t)(b * H_ + h) * S_ * D_;
  const _Float16* Vp = Vt + (size_t)(b * H_ + h) * D_ * S_;
  const float slope = slopes[h];
  const float s16 = slope * 16.0f, s32 = slope * 32.0f;

  v16h qa0 = load_a_frag(Qp + (size_t)q0 * D_, D_, lane);        // d 0..31 (pre-scaled)
  v16h qa1 = load_a_frag(Qp + (size_t)q0 * D_ + 32, D_, lane);   // d 32..63

  float mrow[8], lrow[8], biasr[8];
  v8f o0 = {}, o1 = {}, o2 = {}, o3 = {};
  #pragma unroll
  for (int r = 0; r < 8; ++r) {
    mrow[r] = -1e30f; lrow[r] = 0.f;
    biasr[r] = slope * (float)(jl - (q0 + r + (half << 3)));  // slope*(j-i) at k0=0
  }
  _Float16* ps = pshm[w];

  // Preload K fragments for block 0.
  v16h ka0 = load_b_frag(Kp, D_, lane);
  v16h ka1 = load_b_frag(Kp + 32, D_, lane);
  v16h ka2 = load_b_frag(Kp + (size_t)16 * D_, D_, lane);
  v16h ka3 = load_b_frag(Kp + (size_t)16 * D_ + 32, D_, lane);

  const int nblocks = (q0 + 16 + 31) >> 5;  // causal: only keys <= q0+15
  for (int kb = 0; kb < nblocks; ++kb) {
    const int k0 = kb << 5;
    const int kn = (kb + 1 < nblocks) ? (k0 + 32) : k0;  // clamped prefetch addr
    // Prefetch next block's K fragments (covered by this block's work).
    v16h na0 = load_b_frag(Kp + (size_t)kn * D_, D_, lane);
    v16h na1 = load_b_frag(Kp + (size_t)kn * D_ + 32, D_, lane);
    v16h na2 = load_b_frag(Kp + (size_t)(kn + 16) * D_, D_, lane);
    v16h na3 = load_b_frag(Kp + (size_t)(kn + 16) * D_ + 32, D_, lane);
    // V fragments for this block, issued early.
    v16h vb0 = load_b_frag(Vp + (size_t)(0 * 16) * S_ + k0, S_, lane);
    v16h vb1 = load_b_frag(Vp + (size_t)(1 * 16) * S_ + k0, S_, lane);
    v16h vb2 = load_b_frag(Vp + (size_t)(2 * 16) * S_ + k0, S_, lane);
    v16h vb3 = load_b_frag(Vp + (size_t)(3 * 16) * S_ + k0, S_, lane);

    v8f c0 = {}, c1 = {};
    c0 = wmma_f16(qa0, ka0, c0);
    c0 = wmma_f16(qa1, ka1, c0);
    c1 = wmma_f16(qa0, ka2, c1);
    c1 = wmma_f16(qa1, ka3, c1);

    float alpha[8];
    #pragma unroll
    for (int r = 0; r < 8; ++r) {
      const int i = q0 + r + (half << 3);
      const int j0 = k0 + jl;
      const float b0v = biasr[r];
      float v0 = (j0      <= i) ? c0[r] + b0v       : -1e30f;
      float v1 = (j0 + 16 <= i) ? c1[r] + b0v + s16 : -1e30f;
      const float mx = rowmax16(fmaxf(v0, v1));
      const float mnew = fmaxf(mrow[r], mx);
      const float a = __expf(mrow[r] - mnew);
      const float p0 = __expf(v0 - mnew);
      const float p1 = __expf(v1 - mnew);
      const float rs = rowsum16(p0 + p1);
      lrow[r] = lrow[r] * a + rs;
      mrow[r] = mnew;
      alpha[r] = a;
      biasr[r] = b0v + s32;
      const int row = r + (half << 3);
      ps[row * 40 + jl]      = (_Float16)p0;  // same-type LDS stores (alias-safe)
      ps[row * 40 + 16 + jl] = (_Float16)p1;
    }
    #pragma unroll
    for (int r = 0; r < 8; ++r) {
      o0[r] *= alpha[r]; o1[r] *= alpha[r]; o2[r] *= alpha[r]; o3[r] *= alpha[r];
    }
    // Re-shape P (C layout in LDS) into an A fragment; same-wave LDS is in-order.
    v16h pa;
    {
      const int base = (lane & 15) * 40 + ((lane >> 4) << 3);
      #pragma unroll
      for (int e = 0; e < 8; ++e) pa[e] = ps[base + e];
      #pragma unroll
      for (int e = 0; e < 8; ++e) pa[8 + e] = ps[base + 16 + e];
    }
    o0 = wmma_f16(pa, vb0, o0);
    o1 = wmma_f16(pa, vb1, o1);
    o2 = wmma_f16(pa, vb2, o2);
    o3 = wmma_f16(pa, vb3, o3);

    ka0 = na0; ka1 = na1; ka2 = na2; ka3 = na3;  // rotate prefetched K
  }

  #pragma unroll
  for (int r = 0; r < 8; ++r) {
    const float inv = 1.0f / lrow[r];
    const int m = q0 + r + (half << 3);
    const size_t base = ((size_t)b * S_ + m) * E_ + h * D_ + jl;
    attnh[base + 0]  = (_Float16)(o0[r] * inv);
    attnh[base + 16] = (_Float16)(o1[r] * inv);
    attnh[base + 32] = (_Float16)(o2[r] * inv);
    attnh[base + 48] = (_Float16)(o3[r] * inv);
  }
}

// ---------------------------------------------------------------------------
extern "C" void kernel_launch(void* const* d_in, const int* in_sizes, int n_in,
                              void* d_out, int out_size, void* d_ws, size_t ws_size,
                              hipStream_t stream) {
  const float* x      = (const float*)d_in[0];
  const float* W_kqv  = (const float*)d_in[1];
  const float* b_kqv  = (const float*)d_in[2];
  const float* W_proj = (const float*)d_in[3];
  const float* b_proj = (const float*)d_in[4];
  const float* slopes = (const float*)d_in[5];
  float* out = (float*)d_out;

  char* ws = (char*)d_ws;
  _Float16* xh     = (_Float16*)ws; ws += (size_t)MR_ * E_ * 2;          // 8 MB
  _Float16* Wkqvt  = (_Float16*)ws; ws += (size_t)NKQV * E_ * 2;         // 6 MB
  _Float16* Wprojt = (_Float16*)ws; ws += (size_t)E_ * E_ * 2;           // 2 MB
  _Float16* Qh     = (_Float16*)ws; ws += (size_t)B_ * H_ * S_ * D_ * 2; // 8 MB
  _Float16* Kh     = (_Float16*)ws; ws += (size_t)B_ * H_ * S_ * D_ * 2; // 8 MB
  _Float16* Vt     = (_Float16*)ws; ws += (size_t)B_ * H_ * S_ * D_ * 2; // 8 MB
  _Float16* attnh  = (_Float16*)ws; ws += (size_t)MR_ * E_ * 2;          // 8 MB

  {
    int n = MR_ * E_;
    cvt_f32_f16<<<(n + 255) / 256, 256, 0, stream>>>(x, xh, n);
  }
  transpose_cvt<<<dim3(NKQV / 32, E_ / 32), dim3(32, 8), 0, stream>>>(W_kqv, Wkqvt, E_, NKQV);
  transpose_cvt<<<dim3(E_ / 32, E_ / 32), dim3(32, 8), 0, stream>>>(W_proj, Wprojt, E_, E_);

  gemm_f16<1><<<dim3(NKQV / 64, MR_ / 128), 256, 0, stream>>>(
      xh, Wkqvt, b_kqv, MR_, NKQV, E_, nullptr, Kh, Qh, Vt);

  attn_kernel<<<(B_ * H_ * (S_ / 16)) / 8, 256, 0, stream>>>(Qh, Kh, Vt, slopes, attnh);

  gemm_f16<0><<<dim3(E_ / 64, MR_ / 128), 256, 0, stream>>>(
      attnh, Wprojt, b_proj, MR_, E_, E_, out, nullptr, nullptr, nullptr);
}
